// PPaLMBlock_61950608277527
// MI455X (gfx1250) — matrix operand order repaired and checked by reading, not compiled, not run on hardware
//
#include <hip/hip_runtime.h>
#include <hip/hip_bf16.h>
#include <stdint.h>

// ---------------- problem constants ----------------
#define N_TOK      2048
#define HDIM       2048
#define NHEADS     16
#define DHEAD      64
#define AINNER     1024
#define FFINNER    8192
#define FCOLS      17536   // 1024 + 64 + 64 + 8192 + 8192
#define COL_K      1024
#define COL_V      1088
#define COL_FF     1152
#define COL_GATE   9344
#define QSCALE     0.125f  // 64^-0.5

typedef __attribute__((ext_vector_type(16))) __bf16        v16bf;
typedef __attribute__((ext_vector_type(8)))  float         v8f;
typedef __attribute__((ext_vector_type(4)))  unsigned int  u32x4;

union FragA { v16bf v; u32x4 q[2]; };

__device__ __forceinline__ __bf16 f2bf(float f) {
  union { float f; uint32_t u; } c; c.f = f;
  uint32_t u = c.u;
  u += 0x7FFFu + ((u >> 16) & 1u);           // round-to-nearest-even
  union { unsigned short s; __bf16 b; } o; o.s = (unsigned short)(u >> 16);
  return o.b;
}
__device__ __forceinline__ float bf2f(__bf16 b) {
  union { unsigned short s; __bf16 b; } i; i.b = b;
  union { uint32_t u; float f; } o; o.u = ((uint32_t)i.s) << 16;
  return o.f;
}

// ---- CDNA5 async global->LDS copy (ASYNCcnt path), GVS addressing ----
template <int OFF>
__device__ __forceinline__ void async_ld_b128(uint32_t lds_addr, uint32_t voff_bytes,
                                              const void* sbase) {
  asm volatile("global_load_async_to_lds_b128 %0, %1, %2 offset:%3"
               :: "v"(lds_addr), "v"(voff_bytes), "s"(sbase), "i"(OFF)
               : "memory");
}
__device__ __forceinline__ void wait_asynccnt0() {
  asm volatile("s_wait_asynccnt 0x0" ::: "memory");
}

// ---------------- small prep kernels ----------------
__global__ void zero_kernel(float* p) { if (threadIdx.x == 0) *p = 0.0f; }

__global__ void sumsq_kernel(const float* __restrict__ x, float* __restrict__ acc, int n) {
  __shared__ float red[256];
  float s = 0.0f;
  for (int i = blockIdx.x * blockDim.x + threadIdx.x; i < n; i += gridDim.x * blockDim.x) {
    float v = x[i]; s += v * v;
  }
  red[threadIdx.x] = s; __syncthreads();
  for (int st = 128; st > 0; st >>= 1) {
    if ((int)threadIdx.x < st) red[threadIdx.x] += red[threadIdx.x + st];
    __syncthreads();
  }
  if (threadIdx.x == 0) atomicAdd(acc, red[0]);
}

__global__ void norm_cvt_kernel(const float* __restrict__ x, const float* __restrict__ w,
                                const float* __restrict__ acc, __bf16* __restrict__ xn, int n) {
  int i = blockIdx.x * 256 + threadIdx.x;
  if (i >= n) return;
  float ms = *acc * (1.0f / (float)(N_TOK * HDIM));
  float sc = rsqrtf(ms) + 1e-5f;               // faithful: 1/sqrt(ms) + eps
  xn[i] = f2bf(x[i] * sc * w[i & (HDIM - 1)]);
}

__global__ void cvt_bf16_kernel(const float* __restrict__ s, __bf16* __restrict__ d, int n) {
  int i = blockIdx.x * 256 + threadIdx.x;
  if (i < n) d[i] = f2bf(s[i]);
}

// ---------------- bf16 WMMA GEMM: C[MxN] = A[MxK] * B[KxN] ----------------
// 128x128 block tile, 8 waves (4 row x 2 col), wave = 32x64 = 2x4 WMMA tiles.
// A tile staged via async global->LDS b128; B tile staged transposed via regs.
// Double-buffered LDS, one barrier per K-step.
// MODE 0: store bf16 to Cb;  MODE 1: store f32 to Cf;  MODE 2: accumulate f32 into Cf
template <int MODE>
__global__ __launch_bounds__(256) void gemm_bf16_kernel(
    const __bf16* __restrict__ A, const __bf16* __restrict__ B,
    int M, int K, int Nn, __bf16* __restrict__ Cb, float* __restrict__ Cf)
{
  __shared__ __align__(16) __bf16 Asm[2][128 * 32];   // row-major [row][k]
  __shared__ __align__(16) __bf16 Bsm[2][128 * 40];   // transposed [col][k], padded stride 40
  const int t    = threadIdx.x;
  const int tm   = blockIdx.y * 128;
  const int tn   = blockIdx.x * 128;
  const int wid  = t >> 5, lane = t & 31;
  const int wr   = wid >> 1, wc = wid & 1;            // 4x2 wave grid
  const int hi   = lane >> 4, lr = lane & 15;

  v8f acc[2][4] = {};

  // A async staging: thread t -> row r = t/2, cols c..c+15 (two b128s)
  const int ar = t >> 1, ac = (t & 1) * 16;
  const uint32_t avoff = (uint32_t)(((size_t)(tm + ar) * (size_t)K + (size_t)ac) * 2u);
  // B staging: thread t -> k-row br, cols bc..bc+15
  const int br = t >> 3, bc = (t & 7) * 16;

  // ---- prologue: stage k0 = 0 into buffer 0 ----
  {
    uint32_t lds = (uint32_t)(size_t)(&Asm[0][ar * 32 + ac]);
    const void* sb = (const void*)A;                  // k0 = 0
    async_ld_b128<0>(lds, avoff, sb);
    async_ld_b128<16>(lds, avoff, sb);

    const u32x4* gp = reinterpret_cast<const u32x4*>(&B[(size_t)br * Nn + tn + bc]);
    union { u32x4 q[2]; __bf16 h[16]; } ub;
    ub.q[0] = gp[0]; ub.q[1] = gp[1];
#pragma unroll
    for (int j = 0; j < 16; ++j) Bsm[0][(bc + j) * 40 + br] = ub.h[j];
    wait_asynccnt0();
  }
  __syncthreads();

  int buf = 0;
  for (int k0 = 0; k0 < K; k0 += 32) {
    const int nb = buf ^ 1;
    const bool hasNext = (k0 + 32) < K;

    union { u32x4 q[2]; __bf16 h[16]; } ub;
    if (hasNext) {
      // issue async A(k+1) into the other buffer; B(k+1) into registers
      uint32_t lds = (uint32_t)(size_t)(&Asm[nb][ar * 32 + ac]);
      const void* sb = (const void*)(A + (size_t)(k0 + 32));
      async_ld_b128<0>(lds, avoff, sb);
      async_ld_b128<16>(lds, avoff, sb);
      const u32x4* gp =
          reinterpret_cast<const u32x4*>(&B[(size_t)(k0 + 32 + br) * Nn + tn + bc]);
      ub.q[0] = gp[0]; ub.q[1] = gp[1];
    }

    // ---- compute on current buffer ----
    FragA fa[2], fb[4];
#pragma unroll
    for (int i = 0; i < 2; ++i) {                     // A frag: 16x32, row=lr, K chunks by hi
      const __bf16* p = &Asm[buf][(wr * 32 + i * 16 + lr) * 32];
      fa[i].q[0] = *reinterpret_cast<const u32x4*>(p + hi * 16);
      fa[i].q[1] = *reinterpret_cast<const u32x4*>(p + hi * 16 + 8);
    }
#pragma unroll
    for (int j = 0; j < 4; ++j) {                     // B frag: 32x16, col=lr, K halves by hi
      const __bf16* p = &Bsm[buf][(wc * 64 + j * 16 + lr) * 40 + hi * 16];
      fb[j].q[0] = *reinterpret_cast<const u32x4*>(p);
      fb[j].q[1] = *reinterpret_cast<const u32x4*>(p + 8);
    }
#pragma unroll
    for (int i = 0; i < 2; ++i)
#pragma unroll
      for (int j = 0; j < 4; ++j)
        acc[i][j] = __builtin_amdgcn_wmma_f32_16x16x32_bf16(
            false, fa[i].v, false, fb[j].v, (short)0, acc[i][j], false, false);

    if (hasNext) {                                    // scatter B(k+1) transposed
#pragma unroll
      for (int j = 0; j < 16; ++j) Bsm[nb][(bc + j) * 40 + br] = ub.h[j];
      wait_asynccnt0();
    }
    __syncthreads();
    buf = nb;
  }

#pragma unroll
  for (int i = 0; i < 2; ++i)
#pragma unroll
    for (int j = 0; j < 4; ++j)
#pragma unroll
      for (int r = 0; r < 8; ++r) {
        int m = tm + wr * 32 + i * 16 + r + hi * 8;   // C layout: VGPR r -> M = r + 8*hi
        int n = tn + wc * 64 + j * 16 + lr;
        float v = acc[i][j][r];
        size_t idx = (size_t)m * Nn + n;
        if (MODE == 0)      Cb[idx] = f2bf(v);
        else if (MODE == 1) Cf[idx] = v;
        else                Cf[idx] += v;
      }
}

// ---------------- RoPE / SwiGLU epilogues ----------------
__global__ void rope_q_kernel(const __bf16* __restrict__ fused, const float* __restrict__ sp,
                              const float* __restrict__ cp, __bf16* __restrict__ q) {
  int idx = blockIdx.x * 256 + threadIdx.x;          // pair index over N*512
  int n = idx >> 9, pid = idx & 511;
  int hh = pid >> 5, d0 = (pid & 31) * 2;
  size_t base = (size_t)n * FCOLS + hh * 64 + d0;
  float x1 = bf2f(fused[base])     * QSCALE;
  float x2 = bf2f(fused[base + 1]) * QSCALE;
  float sv = sp[n * 64 + d0], cv = cp[n * 64 + d0];
  size_t ob = (size_t)n * AINNER + hh * 64 + d0;
  q[ob]     = f2bf(x1 * cv - x2 * sv);
  q[ob + 1] = f2bf(x2 * cv + x1 * sv);
}

__global__ void rope_k_kernel(const __bf16* __restrict__ fused, const float* __restrict__ sp,
                              const float* __restrict__ cp, __bf16* __restrict__ k) {
  int idx = blockIdx.x * 256 + threadIdx.x;          // pair index over N*32
  int n = idx >> 5, d0 = (idx & 31) * 2;
  size_t base = (size_t)n * FCOLS + COL_K + d0;
  float x1 = bf2f(fused[base]);
  float x2 = bf2f(fused[base + 1]);
  float sv = sp[n * 64 + d0], cv = cp[n * 64 + d0];
  size_t ob = (size_t)n * DHEAD + d0;
  k[ob]     = f2bf(x1 * cv - x2 * sv);
  k[ob + 1] = f2bf(x2 * cv + x1 * sv);
}

__global__ void swiglu_kernel(const __bf16* __restrict__ fused, __bf16* __restrict__ ffa) {
  size_t idx = (size_t)blockIdx.x * 256 + threadIdx.x; // over N*8192
  int n = (int)(idx >> 13), j = (int)(idx & 8191);
  size_t row = (size_t)n * FCOLS;
  float ff = bf2f(fused[row + COL_FF + j]);
  float g  = bf2f(fused[row + COL_GATE + j]);
  float sw = g / (1.0f + __expf(-g));
  ffa[idx] = f2bf(ff * sw);
}

// ---------------- flash multi-query attention (WMMA) ----------------
// grid (16 qblocks of 128, 16 heads); 8 waves/block; wave owns 16 query rows.
__global__ __launch_bounds__(256) void attn_kernel(
    const __bf16* __restrict__ qb, const __bf16* __restrict__ kb,
    const __bf16* __restrict__ fused, __bf16* __restrict__ outb)
{
  __shared__ __align__(16) __bf16 Ks[64 * 72];        // K tile row-major (== B^T layout)
  __shared__ __align__(16) __bf16 Vt[64 * 72];        // V tile transposed [d][key]
  __shared__ __align__(16) __bf16 Ps[8 * 16 * 72];    // per-wave P staging
  const int t    = threadIdx.x;
  const int wid  = t >> 5, lane = t & 31;
  const int hi   = lane >> 4, lr = lane & 15;
  const int h    = blockIdx.y;
  const int qwbase = blockIdx.x * 128 + wid * 16;

  FragA fq[2];
#pragma unroll
  for (int kk = 0; kk < 2; ++kk) {                    // Q rows direct from global as A-frags
    const __bf16* p = qb + (size_t)(qwbase + lr) * AINNER + h * 64 + kk * 32;
    fq[kk].q[0] = *reinterpret_cast<const u32x4*>(p + hi * 16);
    fq[kk].q[1] = *reinterpret_cast<const u32x4*>(p + hi * 16 + 8);
  }

  float mrow[8], lrow[8];
  v8f of[4] = {};
#pragma unroll
  for (int r = 0; r < 8; ++r) { mrow[r] = -1e30f; lrow[r] = 0.0f; }

  const int kr = t >> 2, dc = (t & 3) * 16;

  for (int kb0 = 0; kb0 < N_TOK; kb0 += 64) {
    {   // K tile: row-major copy
      const u32x4* g = reinterpret_cast<const u32x4*>(kb + (size_t)(kb0 + kr) * DHEAD + dc);
      u32x4* d = reinterpret_cast<u32x4*>(&Ks[kr * 72 + dc]);
      d[0] = g[0]; d[1] = g[1];
    }
    {   // V tile: transpose into Vt[d][key]
      const u32x4* g = reinterpret_cast<const u32x4*>(fused + (size_t)(kb0 + kr) * FCOLS + COL_V + dc);
      union { u32x4 q[2]; __bf16 hh[16]; } uv;
      uv.q[0] = g[0]; uv.q[1] = g[1];
#pragma unroll
      for (int j = 0; j < 16; ++j) Vt[(dc + j) * 72 + kr] = uv.hh[j];
    }
    __syncthreads();

    // S = Q * K^T  (16 q-rows x 64 keys)
    v8f s[4];
#pragma unroll
    for (int jj = 0; jj < 4; ++jj) {
      const __bf16* p = &Ks[(jj * 16 + lr) * 72];
      FragA fb0, fb1;
      fb0.q[0] = *reinterpret_cast<const u32x4*>(p + hi * 16);
      fb0.q[1] = *reinterpret_cast<const u32x4*>(p + hi * 16 + 8);
      fb1.q[0] = *reinterpret_cast<const u32x4*>(p + 32 + hi * 16);
      fb1.q[1] = *reinterpret_cast<const u32x4*>(p + 32 + hi * 16 + 8);
      v8f a = {};
      a = __builtin_amdgcn_wmma_f32_16x16x32_bf16(false, fq[0].v, false, fb0.v, (short)0, a, false, false);
      a = __builtin_amdgcn_wmma_f32_16x16x32_bf16(false, fq[1].v, false, fb1.v, (short)0, a, false, false);
      s[jj] = a;
    }

    // mask (faithful +1e9 on non-causal) + online softmax
#pragma unroll
    for (int r = 0; r < 8; ++r) {
      const int qi = qwbase + r + hi * 8;
      float bm = -1e30f;
#pragma unroll
      for (int jj = 0; jj < 4; ++jj) {
        int key = kb0 + jj * 16 + lr;
        float v = s[jj][r];
        if (key > qi) v = 1e9f;
        s[jj][r] = v;
        bm = fmaxf(bm, v);
      }
#pragma unroll
      for (int off = 1; off < 16; off <<= 1) bm = fmaxf(bm, __shfl_xor(bm, off, 16));
      float mnew  = fmaxf(mrow[r], bm);
      float alpha = __expf(mrow[r] - mnew);
      float psum  = 0.0f;
#pragma unroll
      for (int jj = 0; jj < 4; ++jj) {
        float pv = __expf(s[jj][r] - mnew);
        Ps[wid * 1152 + (r + hi * 8) * 72 + jj * 16 + lr] = f2bf(pv);
        psum += pv;
      }
#pragma unroll
      for (int off = 1; off < 16; off <<= 1) psum += __shfl_xor(psum, off, 16);
      lrow[r] = lrow[r] * alpha + psum;
      mrow[r] = mnew;
#pragma unroll
      for (int dd = 0; dd < 4; ++dd) of[dd][r] *= alpha;
    }

    // O += P * V   (P reloaded as A-frags; same-wave DS is in-order)
    FragA fp[2];
#pragma unroll
    for (int kk2 = 0; kk2 < 2; ++kk2) {
      const __bf16* p = &Ps[wid * 1152 + lr * 72 + kk2 * 32];
      fp[kk2].q[0] = *reinterpret_cast<const u32x4*>(p + hi * 16);
      fp[kk2].q[1] = *reinterpret_cast<const u32x4*>(p + hi * 16 + 8);
    }
#pragma unroll
    for (int dd = 0; dd < 4; ++dd) {
      const __bf16* p = &Vt[(dd * 16 + lr) * 72];
      FragA fv0, fv1;
      fv0.q[0] = *reinterpret_cast<const u32x4*>(p + hi * 16);
      fv0.q[1] = *reinterpret_cast<const u32x4*>(p + hi * 16 + 8);
      fv1.q[0] = *reinterpret_cast<const u32x4*>(p + 32 + hi * 16);
      fv1.q[1] = *reinterpret_cast<const u32x4*>(p + 32 + hi * 16 + 8);
      of[dd] = __builtin_amdgcn_wmma_f32_16x16x32_bf16(false, fp[0].v, false, fv0.v, (short)0, of[dd], false, false);
      of[dd] = __builtin_amdgcn_wmma_f32_16x16x32_bf16(false, fp[1].v, false, fv1.v, (short)0, of[dd], false, false);
    }
    __syncthreads();
  }

#pragma unroll
  for (int dd = 0; dd < 4; ++dd)
#pragma unroll
    for (int r = 0; r < 8; ++r) {
      int qi = qwbase + r + hi * 8;
      float v = of[dd][r] * (1.0f / lrow[r]);
      outb[(size_t)qi * AINNER + h * 64 + dd * 16 + lr] = f2bf(v);
    }
}

// ---------------- host launcher ----------------
extern "C" void kernel_launch(void* const* d_in, const int* in_sizes, int n_in,
                              void* d_out, int out_size, void* d_ws, size_t ws_size,
                              hipStream_t stream) {
  (void)in_sizes; (void)n_in; (void)out_size; (void)ws_size;
  const float* x    = (const float*)d_in[0];
  const float* wi   = (const float*)d_in[1];
  const float* awo  = (const float*)d_in[2];
  const float* fwo  = (const float*)d_in[3];
  const float* nw   = (const float*)d_in[4];
  const float* sinp = (const float*)d_in[5];
  const float* cosp = (const float*)d_in[6];
  float* out = (float*)d_out;

  char* p = (char*)d_ws;
  float*  acc   = (float*)p;   p += 256;
  __bf16* xn    = (__bf16*)p;  p += (size_t)N_TOK * HDIM * 2;
  __bf16* wib   = (__bf16*)p;  p += (size_t)HDIM * FCOLS * 2;
  __bf16* awob  = (__bf16*)p;  p += (size_t)AINNER * HDIM * 2;
  __bf16* fwob  = (__bf16*)p;  p += (size_t)FFINNER * HDIM * 2;
  __bf16* fused = (__bf16*)p;  p += (size_t)N_TOK * FCOLS * 2;
  __bf16* qrot  = (__bf16*)p;  p += (size_t)N_TOK * AINNER * 2;
  __bf16* krot  = (__bf16*)p;  p += (size_t)N_TOK * DHEAD * 2;
  __bf16* ffact = (__bf16*)p;  p += (size_t)N_TOK * FFINNER * 2;
  __bf16* attno = (__bf16*)p;  p += (size_t)N_TOK * AINNER * 2;

  const int nx = N_TOK * HDIM;

  zero_kernel<<<1, 64, 0, stream>>>(acc);
  sumsq_kernel<<<2048, 256, 0, stream>>>(x, acc, nx);
  norm_cvt_kernel<<<nx / 256, 256, 0, stream>>>(x, nw, acc, xn, nx);

  cvt_bf16_kernel<<<(HDIM * FCOLS + 255) / 256, 256, 0, stream>>>(wi, wib, HDIM * FCOLS);
  cvt_bf16_kernel<<<(AINNER * HDIM + 255) / 256, 256, 0, stream>>>(awo, awob, AINNER * HDIM);
  cvt_bf16_kernel<<<(FFINNER * HDIM + 255) / 256, 256, 0, stream>>>(fwo, fwob, FFINNER * HDIM);

  // fused = xn @ wi  (2048x2048 @ 2048x17536), bf16 out
  gemm_bf16_kernel<0><<<dim3(FCOLS / 128, N_TOK / 128), 256, 0, stream>>>(
      xn, wib, N_TOK, HDIM, FCOLS, fused, nullptr);

  rope_q_kernel<<<(N_TOK * 512) / 256, 256, 0, stream>>>(fused, sinp, cosp, qrot);
  rope_k_kernel<<<(N_TOK * 32) / 256, 256, 0, stream>>>(fused, sinp, cosp, krot);
  swiglu_kernel<<<(N_TOK * FFINNER) / 256, 256, 0, stream>>>(fused, ffact);

  attn_kernel<<<dim3(N_TOK / 128, NHEADS), 256, 0, stream>>>(qrot, krot, fused, attno);

  // out  = attno @ attn_wo   (f32 overwrite)
  gemm_bf16_kernel<1><<<dim3(HDIM / 128, N_TOK / 128), 256, 0, stream>>>(
      attno, awob, N_TOK, AINNER, HDIM, nullptr, out);
  // out += ffact @ ff_wo     (f32 accumulate)
  gemm_bf16_kernel<2><<<dim3(HDIM / 128, N_TOK / 128), 256, 0, stream>>>(
      ffact, fwob, N_TOK, FFINNER, HDIM, nullptr, out);
}
